// MoELayer_1855425872528
// MI455X (gfx1250) — matrix operand order repaired and checked by reading, not compiled, not run on hardware
//
#include <hip/hip_runtime.h>
#include <hip/hip_bf16.h>
#include <cstdint>

// Problem constants (B=4, S=2048, H=1024, I=4096, E=8, K=2)
#define TOK   8192
#define HDIM  1024
#define IDIM  4096
#define NEXP  8
#define LDA   40   // LDS row stride in bf16 elems (80B -> 16B aligned rows)

typedef __bf16 bf16;
typedef __attribute__((ext_vector_type(16))) __bf16 v16bf;
typedef __attribute__((ext_vector_type(8)))  __bf16 v8bf;
typedef __attribute__((ext_vector_type(2)))  __bf16 v2bf;
typedef __attribute__((ext_vector_type(8)))  float  v8f;

__device__ __forceinline__ v8f vzero() {
  v8f z;
#pragma unroll
  for (int i = 0; i < 8; ++i) z[i] = 0.f;
  return z;
}

// Packed f32->bf16 pair; bit-cast of a v2bf lowers to v_cvt_pk_bf16_f32.
__device__ __forceinline__ uint32_t pk2(float x, float y) {
  v2bf t;
  t[0] = (__bf16)x;
  t[1] = (__bf16)y;
  return __builtin_bit_cast(uint32_t, t);
}

// Fast SiLU: x * sigmoid(x) with v_rcp_f32 instead of IEEE divide
// (result is quantized to bf16 anyway).
__device__ __forceinline__ float fast_silu(float x) {
  return x * __builtin_amdgcn_rcpf(1.f + __expf(-x));
}

// CDNA5 async DMA: global -> LDS, 16B per lane, tracked on ASYNCcnt.
__device__ __forceinline__ void async_copy16(uint32_t lds_off, const void* gptr) {
  asm volatile("global_load_async_to_lds_b128 %0, %1, off"
               :: "v"(lds_off), "v"(gptr)
               : "memory");
}
__device__ __forceinline__ void wait_async0() {
  asm volatile("s_wait_asynccnt 0" ::: "memory");
}

// A-matrix 16x32 bf16 fragment from LDS tile stored [m][k], row stride LDA.
// lane<16: row=lane, k=0..7 & 16..23 ; lane>=16: row=lane-16, k=8..15 & 24..31
__device__ __forceinline__ v16bf frag_a(const bf16* base, int lane) {
  int r = lane & 15, kg = lane >> 4;
  const bf16* p = base + r * LDA + kg * 8;
  v8bf lo = *(const v8bf*)p;
  v8bf hi = *(const v8bf*)(p + 16);
  v16bf f;
#pragma unroll
  for (int i = 0; i < 8; ++i) { f[i] = lo[i]; f[i + 8] = hi[i]; }
  return f;
}

// B-matrix 32x16 bf16 fragment from LDS tile stored [n][k], row stride LDA.
// lanes 0-15 cover k=0..7 & 16..23 (n=lane), lanes 16-31 cover k=8..15 & 24..31 (n=lane-16)
__device__ __forceinline__ v16bf frag_b(const bf16* base, int lane) {
  int n = lane & 15, kb = (lane >> 4) * 8;
  const bf16* p = base + n * LDA + kb;
  v8bf lo = *(const v8bf*)p;
  v8bf hi = *(const v8bf*)(p + 16);
  v16bf f;
#pragma unroll
  for (int i = 0; i < 8; ++i) { f[i] = lo[i]; f[i + 8] = hi[i]; }
  return f;
}

#define WMMA_BF16(A, Bm, C) \
  __builtin_amdgcn_wmma_f32_16x16x32_bf16(false, (A), false, (Bm), (short)0, (C), false, false)

// -------------------- Router: softmax -> top-2 -> compacted expert lists ---
__global__ void router_kernel(const float* __restrict__ hs,
                              const float* __restrict__ rw,
                              int* __restrict__ counts,
                              int* __restrict__ tokIdx,
                              float* __restrict__ tokW) {
  const int wave = threadIdx.x >> 5, lane = threadIdx.x & 31;
  const int t = blockIdx.x * 8 + wave;
  float acc[NEXP];
#pragma unroll
  for (int e = 0; e < NEXP; ++e) acc[e] = 0.f;
  for (int h = lane; h < HDIM; h += 32) {
    float x = hs[(size_t)t * HDIM + h];
#pragma unroll
    for (int e = 0; e < NEXP; ++e) acc[e] += x * rw[e * HDIM + h];
  }
#pragma unroll
  for (int off = 16; off > 0; off >>= 1) {
#pragma unroll
    for (int e = 0; e < NEXP; ++e) acc[e] += __shfl_xor(acc[e], off, 32);
  }
  if (lane == 0) {
    float mx = acc[0];
#pragma unroll
    for (int e = 1; e < NEXP; ++e) mx = fmaxf(mx, acc[e]);
    float p[NEXP];
#pragma unroll
    for (int e = 0; e < NEXP; ++e) p[e] = __expf(acc[e] - mx);
    // top-2 (first index wins ties, matching lax.top_k); softmax denom cancels
    float b0v = -1.f; int b0i = 0;
#pragma unroll
    for (int e = 0; e < NEXP; ++e) if (p[e] > b0v) { b0v = p[e]; b0i = e; }
    float b1v = -1.f; int b1i = 0;
#pragma unroll
    for (int e = 0; e < NEXP; ++e) if (e != b0i && p[e] > b1v) { b1v = p[e]; b1i = e; }
    float inv = 1.f / (b0v + b1v);
    int p0 = atomicAdd(&counts[b0i], 1);
    tokIdx[b0i * TOK + p0] = t; tokW[b0i * TOK + p0] = b0v * inv;
    int p1 = atomicAdd(&counts[b1i], 1);
    tokIdx[b1i * TOK + p1] = t; tokW[b1i * TOK + p1] = b1v * inv;
  }
}

// ---- Per-expert gathered GEMM: act = bf16( silu(h@gateT) * (h@upT) ) -------
// Tile: 128(M gathered tokens) x 64(N of I), K over H in steps of 32.
__global__ __launch_bounds__(256) void gateup_kernel(
    const float* __restrict__ hs, const float* __restrict__ gw,
    const float* __restrict__ uw, const int* __restrict__ counts,
    const int* __restrict__ tokIdx, bf16* __restrict__ act, int e) {
  __shared__ bf16 sA[128 * LDA];
  __shared__ bf16 sBg[64 * LDA];
  __shared__ bf16 sBu[64 * LDA];
  __shared__ int  sTok[128];

  const int cnt = counts[e];
  const int m0 = blockIdx.x * 128;
  if (m0 >= cnt) return;
  const int n0 = blockIdx.y * 64;
  const int tid = threadIdx.x, lane = tid & 31;
  const int wm = (tid >> 5) & 3, wn = tid >> 7;

  for (int i = tid; i < 128; i += 256) {
    int g = m0 + i;
    sTok[i] = (g < cnt) ? tokIdx[e * TOK + g] : 0;  // invalid rows -> row 0 (masked later)
  }
  __syncthreads();

  // Hoist loop-invariant staging addresses out of the K loop.
  const float* aSrc[4];
  int          aOff[4];
#pragma unroll
  for (int it = 0; it < 4; ++it) {
    int lin = tid + it * 256;
    int row = lin >> 3, c4 = (lin & 7) * 4;
    aSrc[it] = hs + (size_t)sTok[row] * HDIM + c4;
    aOff[it] = row * LDA + c4;
  }
  const float* gSrc[2];
  const float* uSrc[2];
  int          bOff[2];
  const float* gbase = gw + (size_t)e * IDIM * HDIM;
  const float* ubase = uw + (size_t)e * IDIM * HDIM;
#pragma unroll
  for (int it = 0; it < 2; ++it) {
    int lin = tid + it * 256;
    int row = lin >> 3, c4 = (lin & 7) * 4;
    gSrc[it] = gbase + (size_t)(n0 + row) * HDIM + c4;
    uSrc[it] = ubase + (size_t)(n0 + row) * HDIM + c4;
    bOff[it] = row * LDA + c4;
  }

  v8f gAcc[2][2], uAcc[2][2];
#pragma unroll
  for (int i = 0; i < 2; ++i)
#pragma unroll
    for (int j = 0; j < 2; ++j) { gAcc[i][j] = vzero(); uAcc[i][j] = vzero(); }

  for (int k0 = 0; k0 < HDIM; k0 += 32) {
    __syncthreads();
#pragma unroll
    for (int it = 0; it < 4; ++it) {
      float4 v = *(const float4*)(aSrc[it] + k0);
      *(uint2*)(sA + aOff[it]) = make_uint2(pk2(v.x, v.y), pk2(v.z, v.w));
    }
#pragma unroll
    for (int it = 0; it < 2; ++it) {
      float4 v = *(const float4*)(gSrc[it] + k0);
      *(uint2*)(sBg + bOff[it]) = make_uint2(pk2(v.x, v.y), pk2(v.z, v.w));
      float4 w = *(const float4*)(uSrc[it] + k0);
      *(uint2*)(sBu + bOff[it]) = make_uint2(pk2(w.x, w.y), pk2(w.z, w.w));
    }
    __syncthreads();

    v16bf a0  = frag_a(sA  + (wm * 32) * LDA,      lane);
    v16bf a1  = frag_a(sA  + (wm * 32 + 16) * LDA, lane);
    v16bf bg0 = frag_b(sBg + (wn * 32) * LDA,      lane);
    v16bf bg1 = frag_b(sBg + (wn * 32 + 16) * LDA, lane);
    v16bf bu0 = frag_b(sBu + (wn * 32) * LDA,      lane);
    v16bf bu1 = frag_b(sBu + (wn * 32 + 16) * LDA, lane);

    gAcc[0][0] = WMMA_BF16(a0, bg0, gAcc[0][0]);
    gAcc[0][1] = WMMA_BF16(a0, bg1, gAcc[0][1]);
    gAcc[1][0] = WMMA_BF16(a1, bg0, gAcc[1][0]);
    gAcc[1][1] = WMMA_BF16(a1, bg1, gAcc[1][1]);
    uAcc[0][0] = WMMA_BF16(a0, bu0, uAcc[0][0]);
    uAcc[0][1] = WMMA_BF16(a0, bu1, uAcc[0][1]);
    uAcc[1][0] = WMMA_BF16(a1, bu0, uAcc[1][0]);
    uAcc[1][1] = WMMA_BF16(a1, bu1, uAcc[1][1]);
  }

  // Fused SiLU(g)*u epilogue -> bf16 act. Fast path: full tile, no guards.
  const int n16 = lane & 15, mh = (lane >> 4) * 8;
  auto epilogue = [&](bool guard) {
#pragma unroll
    for (int im = 0; im < 2; ++im)
#pragma unroll
      for (int r = 0; r < 8; ++r) {
        int pos = m0 + wm * 32 + im * 16 + mh + r;
        if (!guard || pos < cnt) {
          bf16* dst = act + (size_t)pos * IDIM + (n0 + wn * 32 + n16);
#pragma unroll
          for (int jn = 0; jn < 2; ++jn) {
            float a = fast_silu(gAcc[im][jn][r]) * uAcc[im][jn][r];
            dst[jn * 16] = (bf16)a;
          }
        }
      }
  };
  if (m0 + 128 <= cnt) epilogue(false); else epilogue(true);
}

// ---- Per-expert gathered GEMM: out[tok] += w * (act @ down_wT) -------------
__global__ __launch_bounds__(256) void down_kernel(
    const bf16* __restrict__ act, const float* __restrict__ dw,
    const int* __restrict__ counts, const int* __restrict__ tokIdx,
    const float* __restrict__ tokW, float* __restrict__ out, int e) {
  __shared__ bf16  sA[128 * LDA];
  __shared__ bf16  sB[64 * LDA];
  __shared__ int   sTok[128];
  __shared__ float sW[128];

  const int cnt = counts[e];
  const int m0 = blockIdx.x * 128;
  if (m0 >= cnt) return;
  const int n0 = blockIdx.y * 64;
  const int tid = threadIdx.x, lane = tid & 31;
  const int wm = (tid >> 5) & 3, wn = tid >> 7;

  for (int i = tid; i < 128; i += 256) {
    int g = m0 + i;
    sTok[i] = (g < cnt) ? tokIdx[e * TOK + g] : -1;
    sW[i]   = (g < cnt) ? tokW[e * TOK + g] : 0.f;
  }

  // Hoisted staging addresses: A is a pure bf16 copy -> async DMA to LDS.
  const bf16* aSrc[2];
  uint32_t    aLds[2];
#pragma unroll
  for (int it = 0; it < 2; ++it) {
    int lin = tid + it * 256;
    int row = lin >> 2, c8 = (lin & 3) * 8;
    aSrc[it] = act + (size_t)(m0 + row) * IDIM + c8;
    aLds[it] = (uint32_t)(uintptr_t)(sA + row * LDA + c8);
  }
  const float* bSrc[2];
  int          bOff[2];
  const float* dbase = dw + (size_t)e * HDIM * IDIM;
#pragma unroll
  for (int it = 0; it < 2; ++it) {
    int lin = tid + it * 256;
    int row = lin >> 3, c4 = (lin & 7) * 4;
    bSrc[it] = dbase + (size_t)(n0 + row) * IDIM + c4;
    bOff[it] = row * LDA + c4;
  }

  v8f acc[2][2];
#pragma unroll
  for (int i = 0; i < 2; ++i)
#pragma unroll
    for (int j = 0; j < 2; ++j) acc[i][j] = vzero();

  for (int k0 = 0; k0 < IDIM; k0 += 32) {
    __syncthreads();
#pragma unroll
    for (int it = 0; it < 2; ++it)
      async_copy16(aLds[it], aSrc[it] + k0);   // ASYNCcnt-tracked DMA, no VGPR roundtrip
#pragma unroll
    for (int it = 0; it < 2; ++it) {
      float4 v = *(const float4*)(bSrc[it] + k0);
      *(uint2*)(sB + bOff[it]) = make_uint2(pk2(v.x, v.y), pk2(v.z, v.w));
    }
    wait_async0();
    __syncthreads();

    v16bf a0 = frag_a(sA + (wm * 32) * LDA,      lane);
    v16bf a1 = frag_a(sA + (wm * 32 + 16) * LDA, lane);
    v16bf b0 = frag_b(sB + (wn * 32) * LDA,      lane);
    v16bf b1 = frag_b(sB + (wn * 32 + 16) * LDA, lane);

    acc[0][0] = WMMA_BF16(a0, b0, acc[0][0]);
    acc[0][1] = WMMA_BF16(a0, b1, acc[0][1]);
    acc[1][0] = WMMA_BF16(a1, b0, acc[1][0]);
    acc[1][1] = WMMA_BF16(a1, b1, acc[1][1]);
  }

  // Scatter-accumulate. Fast path: full tile, no per-element guards.
  const int n16 = lane & 15, mh = (lane >> 4) * 8;
  auto epilogue = [&](bool guard) {
#pragma unroll
    for (int im = 0; im < 2; ++im)
#pragma unroll
      for (int r = 0; r < 8; ++r) {
        int lp = wm * 32 + im * 16 + mh + r;
        if (!guard || (m0 + lp) < cnt) {
          int tok = sTok[lp];
          float w = sW[lp];
          float* dst = out + (size_t)tok * HDIM + (n0 + wn * 32 + n16);
#pragma unroll
          for (int jn = 0; jn < 2; ++jn)
            dst[jn * 16] += w * acc[im][jn][r];
        }
      }
  };
  if (m0 + 128 <= cnt) epilogue(false); else epilogue(true);
}

extern "C" void kernel_launch(void* const* d_in, const int* in_sizes, int n_in,
                              void* d_out, int out_size, void* d_ws, size_t ws_size,
                              hipStream_t stream) {
  (void)in_sizes; (void)n_in; (void)ws_size;
  const float* hs = (const float*)d_in[0];
  const float* rw = (const float*)d_in[1];
  const float* gw = (const float*)d_in[2];
  const float* uw = (const float*)d_in[3];
  const float* dw = (const float*)d_in[4];
  float* out = (float*)d_out;

  // Workspace layout: [counts 256B][tokIdx E*T*4][tokW E*T*4][act T*I*2]
  char*  ws     = (char*)d_ws;
  int*   counts = (int*)ws;
  int*   tokIdx = (int*)(ws + 256);
  float* tokW   = (float*)(ws + 256 + (size_t)NEXP * TOK * sizeof(int));
  bf16*  act    = (bf16*)(ws + 256 + 2 * (size_t)NEXP * TOK * sizeof(int));

  hipMemsetAsync(counts, 0, NEXP * sizeof(int), stream);
  hipMemsetAsync(d_out, 0, (size_t)out_size * sizeof(float), stream);

  router_kernel<<<TOK / 8, 256, 0, stream>>>(hs, rw, counts, tokIdx, tokW);

  for (int e = 0; e < NEXP; ++e) {
    gateup_kernel<<<dim3(TOK / 128, IDIM / 64), 256, 0, stream>>>(
        hs, gw, uw, counts, tokIdx, act, e);
    down_kernel<<<dim3(TOK / 128, HDIM / 64), 256, 0, stream>>>(
        act, dw, counts, tokIdx, tokW, out, e);
  }
}
